// LongformerSegmentClassifier_61864708932287
// MI455X (gfx1250) — compile-verified, matrix operand order checked by reading, not run on hardware
//
#include <hip/hip_runtime.h>
#include <math.h>

typedef __attribute__((ext_vector_type(16))) __bf16 v16bf;
typedef __attribute__((ext_vector_type(8)))  float  v8f;

#define S_LEN 4096
#define DMODEL 768
#define NHEAD 12
#define DHEAD 64
#define WWIN 256
#define NBLK 16
#define FFDIM 3072
#define NLAYER 12
#define NEGINF (-1e9f)

union BFfrag { v16bf v; unsigned u[8]; };

__device__ __forceinline__ unsigned short f2bf(float f) {
  union { float f; unsigned u; } t; t.f = f;
  unsigned u = t.u + 0x7fffu + ((t.u >> 16) & 1u);  // round-to-nearest-even
  return (unsigned short)(u >> 16);
}

// k-pair base for 16-bit A/B fragments (ISA 7.12.2): VGPR v holds K=(kp,kp+1)
__device__ __forceinline__ int kpair(int v, int half) {
  return ((v < 4) ? (2 * v) : (16 + 2 * (v - 4))) + 8 * half;
}

// ---------------------------------------------------------------------------
// fp32 -> bf16 conversion, 8 elements per thread. n must be a multiple of 2048.
// ---------------------------------------------------------------------------
__global__ __launch_bounds__(256) void cvt_bf16(
    const float* __restrict__ in, unsigned short* __restrict__ out, int n) {
  int i = (blockIdx.x * 256 + threadIdx.x) * 8;
  if (i + 8 <= n) {
#pragma unroll
    for (int j = 0; j < 8; ++j) out[i + j] = f2bf(in[i + j]);
  }
}

// ---------------------------------------------------------------------------
// Tiled GEMM on pre-converted bf16 operands:
//   C[M,N] = act( (A[M,K] @ B[K,N] + bias[N]) * alpha ),  C fp32
// Block tile 128x128x32, 512 threads = 16 waves in a 4x4 grid; each wave owns
// a 32x32 C tile = 4 WMMA accumulators (4 v_wmma per K-step). Next K-slab is
// prefetched into registers while the current one is consumed.
// M % 128 == 0, N % 128 == 0, K % 32 == 0 (true for all uses).
// act: 0 = none, 1 = exact GELU
// ---------------------------------------------------------------------------
__global__ __launch_bounds__(512) void gemm_bf16(
    const unsigned short* __restrict__ A, const unsigned short* __restrict__ B,
    const float* __restrict__ bias, float* __restrict__ C,
    int M, int N, int K, float alpha, int act) {
  __shared__ unsigned short As[128][40];  // row stride 80B (16B aligned)
  __shared__ unsigned short Bt[128][40];  // B transposed: [n][k]
  const int tid = threadIdx.x;
  const int lane = tid & 31, wave = tid >> 5;
  const int half = lane >> 4, mr = lane & 15;
  const int waveM = wave >> 2, waveN = wave & 3;
  const int bm = blockIdx.y * 128, bn = blockIdx.x * 128;

  v8f acc[2][2];
  { v8f z = {}; acc[0][0] = z; acc[0][1] = z; acc[1][0] = z; acc[1][1] = z; }

  const int arow = tid >> 2, acol = (tid & 3) * 8;    // 128 x 32 A tile
  const int brow = tid >> 4, bcol = (tid & 15) * 8;   // 32 x 128 B tile

  const unsigned short* apg = A + (size_t)(bm + arow) * K + acol;
  const unsigned short* bpg = B + (size_t)brow * N + bn + bcol;

  uint4 ra = *(const uint4*)apg;
  uint4 rb = *(const uint4*)bpg;

  for (int k0 = 0; k0 < K; k0 += 32) {
    *(uint4*)&As[arow][acol] = ra;
    {
      union { uint4 q; unsigned short s[8]; } ub; ub.q = rb;
#pragma unroll
      for (int j = 0; j < 8; ++j) Bt[bcol + j][brow] = ub.s[j];
    }
    __syncthreads();

    if (k0 + 32 < K) {   // prefetch next K-slab while WMMAs run
      ra = *(const uint4*)(apg + k0 + 32);
      rb = *(const uint4*)(bpg + (size_t)(k0 + 32) * N);
    }

    BFfrag fa0, fa1, fb0, fb1;
#pragma unroll
    for (int v = 0; v < 8; ++v) {
      int kk = kpair(v, half);
      fa0.u[v] = *(const unsigned*)&As[waveM * 32 + mr][kk];
      fa1.u[v] = *(const unsigned*)&As[waveM * 32 + 16 + mr][kk];
      fb0.u[v] = *(const unsigned*)&Bt[waveN * 32 + mr][kk];
      fb1.u[v] = *(const unsigned*)&Bt[waveN * 32 + 16 + mr][kk];
    }
    acc[0][0] = __builtin_amdgcn_wmma_f32_16x16x32_bf16(false, fa0.v, false, fb0.v, (short)0, acc[0][0], false, false);
    acc[0][1] = __builtin_amdgcn_wmma_f32_16x16x32_bf16(false, fa0.v, false, fb1.v, (short)0, acc[0][1], false, false);
    acc[1][0] = __builtin_amdgcn_wmma_f32_16x16x32_bf16(false, fa1.v, false, fb0.v, (short)0, acc[1][0], false, false);
    acc[1][1] = __builtin_amdgcn_wmma_f32_16x16x32_bf16(false, fa1.v, false, fb1.v, (short)0, acc[1][1], false, false);
    __syncthreads();
  }

#pragma unroll
  for (int ns = 0; ns < 2; ++ns) {
    const int n = bn + waveN * 32 + ns * 16 + mr;
    const float bv = bias ? bias[n] : 0.0f;
#pragma unroll
    for (int msub = 0; msub < 2; ++msub) {
#pragma unroll
      for (int r = 0; r < 8; ++r) {
        int m0 = bm + waveM * 32 + msub * 16 + r + 8 * half;
        float v0 = (acc[msub][ns][r] + bv) * alpha;
        if (act == 1) v0 = 0.5f * v0 * (1.0f + erff(v0 * 0.70710678118f));
        C[(size_t)m0 * N + n] = v0;
      }
    }
  }
}

// ---------------------------------------------------------------------------
// Flash-style sliding-window attention. Grid (16 blocks, 12 heads), 256 thr.
// Each wave owns 32 query rows; iterates over 24 key-chunks of 32 keys
// (the 3W=768 local band) with online softmax; then folds in the global
// (pos 0) key column. q is pre-scaled by 1/sqrt(DH).
// q,k,v,ctx layout: [S][H*DH] fp32.
// ---------------------------------------------------------------------------
__global__ __launch_bounds__(256) void attn_window(
    const float* __restrict__ q, const float* __restrict__ k,
    const float* __restrict__ v, const int* __restrict__ mask,
    float* __restrict__ ctx) {
  const int b = blockIdx.x;   // query block
  const int h = blockIdx.y;   // head
  const int tid = threadIdx.x;
  const int lane = tid & 31, w = tid >> 5;
  const int half = lane >> 4, mr = lane & 15;

  __shared__ unsigned short Ks[32][66];      // [key][dh] bf16
  __shared__ unsigned short Vt[64][34];      // [dh][key] bf16 (transposed)
  __shared__ unsigned short Ps[8][32][34];   // per-wave probs bf16
  __shared__ float sg[256];                  // global-key scores per row
  __shared__ float k0s[64], v0s[64];
  __shared__ unsigned char vmask[768];

  if (tid < 64) {
    k0s[tid] = k[(size_t)h * DHEAD + tid];   // row 0
    v0s[tid] = v[(size_t)h * DHEAD + tid];
  }
  for (int j = tid; j < 768; j += 256) {
    int pos = (b - 1) * WWIN + j;
    vmask[j] = (pos >= 0 && pos < S_LEN && pos != 0 && mask[pos] != 0) ? 1 : 0;
  }
  __syncthreads();
  {
    float s = 0.f;
    const float* qp = q + (size_t)(b * WWIN + tid) * DMODEL + h * DHEAD;
#pragma unroll
    for (int d = 0; d < DHEAD; ++d) s += qp[d] * k0s[d];
    sg[tid] = s;
  }
  __syncthreads();

  // Q fragments (2 m-subtiles x 2 k-steps), loaded straight from global
  BFfrag qf[2][2];
#pragma unroll
  for (int ms = 0; ms < 2; ++ms) {
    const float* qp = q + (size_t)(b * WWIN + w * 32 + ms * 16 + mr) * DMODEL + h * DHEAD;
#pragma unroll
    for (int ks = 0; ks < 2; ++ks)
#pragma unroll
      for (int vv = 0; vv < 8; ++vv) {
        int kk = kpair(vv, half) + ks * 32;
        qf[ms][ks].u[vv] =
            (unsigned)f2bf(qp[kk]) | ((unsigned)f2bf(qp[kk + 1]) << 16);
      }
  }

  v8f ctxacc[2][4];
  float rmax[2][8], rsum[2][8];
  {
    v8f z = {};
#pragma unroll
    for (int ms = 0; ms < 2; ++ms) {
#pragma unroll
      for (int t = 0; t < 4; ++t) ctxacc[ms][t] = z;
#pragma unroll
      for (int r = 0; r < 8; ++r) { rmax[ms][r] = -1e30f; rsum[ms][r] = 0.f; }
    }
  }

  const int krow = tid >> 3, kcol = (tid & 7) * 8;
  for (int c = 0; c < 24; ++c) {
    __syncthreads();  // all waves done reading previous K/V tiles
    {
      int pos = (b - 1) * WWIN + c * 32 + krow;
      bool inr = (pos >= 0 && pos < S_LEN);
      const float* kp = k + (size_t)(inr ? pos : 0) * DMODEL + h * DHEAD + kcol;
      const float* vp = v + (size_t)(inr ? pos : 0) * DMODEL + h * DHEAD + kcol;
#pragma unroll
      for (int j = 0; j < 8; ++j) {
        Ks[krow][kcol + j] = f2bf(inr ? kp[j] : 0.f);
        Vt[kcol + j][krow] = f2bf(inr ? vp[j] : 0.f);
      }
    }
    __syncthreads();

    // scores S = Q . K^T  (2 m-subtiles x 2 key col-tiles)
    v8f sacc[2][2];
    { v8f z = {}; sacc[0][0] = z; sacc[0][1] = z; sacc[1][0] = z; sacc[1][1] = z; }
#pragma unroll
    for (int ks = 0; ks < 2; ++ks)
#pragma unroll
      for (int nt = 0; nt < 2; ++nt) {
        BFfrag fb;
#pragma unroll
        for (int vv = 0; vv < 8; ++vv)
          fb.u[vv] = *(const unsigned*)&Ks[nt * 16 + mr][kpair(vv, half) + ks * 32];
        sacc[0][nt] = __builtin_amdgcn_wmma_f32_16x16x32_bf16(
            false, qf[0][ks].v, false, fb.v, (short)0, sacc[0][nt], false, false);
        sacc[1][nt] = __builtin_amdgcn_wmma_f32_16x16x32_bf16(
            false, qf[1][ks].v, false, fb.v, (short)0, sacc[1][nt], false, false);
      }

    // mask + online softmax + stage P in LDS (bf16)
#pragma unroll
    for (int ms = 0; ms < 2; ++ms) {
      float s0a[8], s1a[8], lm[8];
#pragma unroll
      for (int r = 0; r < 8; ++r) {
        int qi = w * 32 + ms * 16 + r + 8 * half;
        int kj0 = c * 32 + mr, kj1 = kj0 + 16;
        int d0 = kj0 - WWIN - qi, d1 = kj1 - WWIN - qi;
        bool ok0 = vmask[kj0] && (d0 <= WWIN) && (d0 >= -WWIN);
        bool ok1 = vmask[kj1] && (d1 <= WWIN) && (d1 >= -WWIN);
        s0a[r] = ok0 ? sacc[ms][0][r] : NEGINF;
        s1a[r] = ok1 ? sacc[ms][1][r] : NEGINF;
        lm[r] = fmaxf(s0a[r], s1a[r]);
      }
#pragma unroll
      for (int r = 0; r < 8; ++r) {
#pragma unroll
        for (int off = 1; off < 16; off <<= 1)
          lm[r] = fmaxf(lm[r], __shfl_xor(lm[r], off, 32));
        float nm = fmaxf(rmax[ms][r], lm[r]);
        float al = __expf(rmax[ms][r] - nm);
        rmax[ms][r] = nm;
        float p0 = __expf(s0a[r] - nm);
        float p1 = __expf(s1a[r] - nm);
        float ls = p0 + p1;
#pragma unroll
        for (int off = 1; off < 16; off <<= 1)
          ls += __shfl_xor(ls, off, 32);
        rsum[ms][r] = rsum[ms][r] * al + ls;
#pragma unroll
        for (int t = 0; t < 4; ++t) ctxacc[ms][t][r] *= al;
        int row = ms * 16 + r + 8 * half;
        Ps[w][row][mr] = f2bf(p0);
        Ps[w][row][mr + 16] = f2bf(p1);
      }
    }
    // cross-lane LDS RAW within the wave: wait for DS stores to land
    asm volatile("s_wait_dscnt 0" ::: "memory");

    // ctx += P(32x32) x V(32x64)
#pragma unroll
    for (int ms = 0; ms < 2; ++ms) {
      BFfrag fa;
#pragma unroll
      for (int vv = 0; vv < 8; ++vv)
        fa.u[vv] = *(const unsigned*)&Ps[w][ms * 16 + mr][kpair(vv, half)];
#pragma unroll
      for (int t = 0; t < 4; ++t) {
        BFfrag fb;
#pragma unroll
        for (int vv = 0; vv < 8; ++vv)
          fb.u[vv] = *(const unsigned*)&Vt[t * 16 + mr][kpair(vv, half)];
        ctxacc[ms][t] = __builtin_amdgcn_wmma_f32_16x16x32_bf16(
            false, fa.v, false, fb.v, (short)0, ctxacc[ms][t], false, false);
      }
    }
  }

  // fold in the global (pos 0) key column
  const int m0v = mask[0];
#pragma unroll
  for (int ms = 0; ms < 2; ++ms)
#pragma unroll
    for (int r = 0; r < 8; ++r) {
      int qi = w * 32 + ms * 16 + r + 8 * half;
      float s = (m0v != 0) ? sg[qi] : NEGINF;
      float nm = fmaxf(rmax[ms][r], s);
      float al = __expf(rmax[ms][r] - nm);
      float p0 = __expf(s - nm);
      rmax[ms][r] = nm;
      rsum[ms][r] = rsum[ms][r] * al + p0;
#pragma unroll
      for (int t = 0; t < 4; ++t)
        ctxacc[ms][t][r] = ctxacc[ms][t][r] * al + p0 * v0s[t * 16 + mr];
    }

  // write ctx / rsum
#pragma unroll
  for (int ms = 0; ms < 2; ++ms)
#pragma unroll
    for (int r = 0; r < 8; ++r) {
      int qi = w * 32 + ms * 16 + r + 8 * half;
      float inv = 1.0f / rsum[ms][r];
      size_t base = (size_t)(b * WWIN + qi) * DMODEL + h * DHEAD;
#pragma unroll
      for (int t = 0; t < 4; ++t) ctx[base + t * 16 + mr] = ctxacc[ms][t][r] * inv;
    }
}

// ---------------------------------------------------------------------------
// Global (CLS) token attention: one block per head, full softmax over S.
// Overwrites ctx row 0.
// ---------------------------------------------------------------------------
__global__ __launch_bounds__(256) void attn_global(
    const float* __restrict__ qg, const float* __restrict__ kg,
    const float* __restrict__ vg, const int* __restrict__ mask,
    float* __restrict__ ctx) {
  const int h = blockIdx.x, tid = threadIdx.x;
  __shared__ float sc[S_LEN];
  __shared__ float qh[DHEAD];
  __shared__ float red[256];
  if (tid < DHEAD) qh[tid] = qg[h * DHEAD + tid];
  __syncthreads();
  float lmax = -1e30f;
  for (int i = tid; i < S_LEN; i += 256) {
    const float* kp = kg + (size_t)i * DMODEL + h * DHEAD;
    float s = 0.f;
#pragma unroll
    for (int d = 0; d < DHEAD; ++d) s += qh[d] * kp[d];
    s = (mask[i] != 0) ? s : NEGINF;
    sc[i] = s;
    lmax = fmaxf(lmax, s);
  }
  red[tid] = lmax; __syncthreads();
  for (int o = 128; o > 0; o >>= 1) {
    if (tid < o) red[tid] = fmaxf(red[tid], red[tid + o]);
    __syncthreads();
  }
  float gmax = red[0]; __syncthreads();
  float lsum = 0.f;
  for (int i = tid; i < S_LEN; i += 256) {
    float p = __expf(sc[i] - gmax);
    sc[i] = p; lsum += p;
  }
  __syncthreads();
  red[tid] = lsum; __syncthreads();
  for (int o = 128; o > 0; o >>= 1) {
    if (tid < o) red[tid] += red[tid + o];
    __syncthreads();
  }
  float inv = 1.0f / red[0];
  if (tid < DHEAD) {
    float a = 0.f;
    for (int i = 0; i < S_LEN; ++i)
      a += sc[i] * vg[(size_t)i * DMODEL + h * DHEAD + tid];
    ctx[h * DHEAD + tid] = a * inv;
  }
}

// out[j] = (x0 . W[:,j] + b[j]) * alpha   (single row x DxN)
__global__ __launch_bounds__(256) void vec_gemm(
    const float* __restrict__ x0, const float* __restrict__ Wm,
    const float* __restrict__ bias, float* __restrict__ out, int N, float alpha) {
  __shared__ float xs[DMODEL];
  const int tid = threadIdx.x;
  for (int d = tid; d < DMODEL; d += 256) xs[d] = x0[d];
  __syncthreads();
  int j = blockIdx.x * 256 + tid;
  if (j < N) {
    float s = 0.f;
    for (int d = 0; d < DMODEL; ++d) s += xs[d] * Wm[(size_t)d * N + j];
    out[j] = (s + bias[j]) * alpha;
  }
}

// out = LN(a + b) * g + beta   (one block per row, D=768 = 3*256)
__global__ __launch_bounds__(256) void add_ln(
    const float* __restrict__ a, const float* __restrict__ b,
    const float* __restrict__ g, const float* __restrict__ beta,
    float* __restrict__ out) {
  const int row = blockIdx.x, tid = threadIdx.x;
  __shared__ float red[256];
  const size_t base = (size_t)row * DMODEL;
  float x[3]; float s = 0.f;
#pragma unroll
  for (int j = 0; j < 3; ++j) {
    int d = tid + j * 256;
    x[j] = a[base + d] + b[base + d];
    s += x[j];
  }
  red[tid] = s; __syncthreads();
  for (int o = 128; o > 0; o >>= 1) { if (tid < o) red[tid] += red[tid + o]; __syncthreads(); }
  float mean = red[0] / DMODEL; __syncthreads();
  float s2 = 0.f;
#pragma unroll
  for (int j = 0; j < 3; ++j) { float dv = x[j] - mean; s2 += dv * dv; }
  red[tid] = s2; __syncthreads();
  for (int o = 128; o > 0; o >>= 1) { if (tid < o) red[tid] += red[tid + o]; __syncthreads(); }
  float rstd = rsqrtf(red[0] / DMODEL + 1e-5f);
#pragma unroll
  for (int j = 0; j < 3; ++j) {
    int d = tid + j * 256;
    out[base + d] = (x[j] - mean) * rstd * g[d] + beta[d];
  }
}

// x = LN(emb_word[ids[row]] + emb_pos[row]) * g + beta
__global__ __launch_bounds__(256) void embed_ln(
    const int* __restrict__ ids, const float* __restrict__ ew,
    const float* __restrict__ ep, const float* __restrict__ g,
    const float* __restrict__ beta, float* __restrict__ out) {
  const int row = blockIdx.x, tid = threadIdx.x;
  __shared__ float red[256];
  const size_t base = (size_t)row * DMODEL;
  const size_t wb = (size_t)ids[row] * DMODEL;
  float x[3]; float s = 0.f;
#pragma unroll
  for (int j = 0; j < 3; ++j) {
    int d = tid + j * 256;
    x[j] = ew[wb + d] + ep[base + d];
    s += x[j];
  }
  red[tid] = s; __syncthreads();
  for (int o = 128; o > 0; o >>= 1) { if (tid < o) red[tid] += red[tid + o]; __syncthreads(); }
  float mean = red[0] / DMODEL; __syncthreads();
  float s2 = 0.f;
#pragma unroll
  for (int j = 0; j < 3; ++j) { float dv = x[j] - mean; s2 += dv * dv; }
  red[tid] = s2; __syncthreads();
  for (int o = 128; o > 0; o >>= 1) { if (tid < o) red[tid] += red[tid + o]; __syncthreads(); }
  float rstd = rsqrtf(red[0] / DMODEL + 1e-5f);
#pragma unroll
  for (int j = 0; j < 3; ++j) {
    int d = tid + j * 256;
    out[base + d] = (x[j] - mean) * rstd * g[d] + beta[d];
  }
}

// logits[0..1] = x_cls . Wc[:,j] + bc[j]
__global__ __launch_bounds__(256) void classifier(
    const float* __restrict__ x, const float* __restrict__ Wc,
    const float* __restrict__ bc, float* __restrict__ out) {
  __shared__ float red[2][256];
  const int tid = threadIdx.x;
  float s0 = 0.f, s1 = 0.f;
  for (int d = tid; d < DMODEL; d += 256) {
    float xv = x[d];
    s0 += xv * Wc[d * 2 + 0];
    s1 += xv * Wc[d * 2 + 1];
  }
  red[0][tid] = s0; red[1][tid] = s1; __syncthreads();
  for (int o = 128; o > 0; o >>= 1) {
    if (tid < o) { red[0][tid] += red[0][tid + o]; red[1][tid] += red[1][tid + o]; }
    __syncthreads();
  }
  if (tid == 0) { out[0] = red[0][0] + bc[0]; out[1] = red[1][0] + bc[1]; }
}

// ---------------------------------------------------------------------------
extern "C" void kernel_launch(void* const* d_in, const int* in_sizes, int n_in,
                              void* d_out, int out_size, void* d_ws, size_t ws_size,
                              hipStream_t stream) {
  (void)in_sizes; (void)n_in; (void)out_size; (void)ws_size;
  const int*   ids  = (const int*)d_in[0];
  const int*   mask = (const int*)d_in[1];
  const float* ew   = (const float*)d_in[2];
  const float* ep   = (const float*)d_in[3];
  const float* eg   = (const float*)d_in[4];
  const float* eb   = (const float*)d_in[5];
  const float* Wq  = (const float*)d_in[6],  *bq  = (const float*)d_in[7];
  const float* Wk  = (const float*)d_in[8],  *bk  = (const float*)d_in[9];
  const float* Wv  = (const float*)d_in[10], *bv  = (const float*)d_in[11];
  const float* Wqg = (const float*)d_in[12], *bqg = (const float*)d_in[13];
  const float* Wkg = (const float*)d_in[14], *bkg = (const float*)d_in[15];
  const float* Wvg = (const float*)d_in[16], *bvg = (const float*)d_in[17];
  const float* Wo  = (const float*)d_in[18], *bo  = (const float*)d_in[19];
  const float* l1g = (const float*)d_in[20], *l1b = (const float*)d_in[21];
  const float* W1  = (const float*)d_in[22], *b1  = (const float*)d_in[23];
  const float* W2  = (const float*)d_in[24], *b2  = (const float*)d_in[25];
  const float* l2g = (const float*)d_in[26], *l2b = (const float*)d_in[27];
  const float* Wc  = (const float*)d_in[28], *bc  = (const float*)d_in[29];

  const size_t SD  = (size_t)S_LEN * DMODEL;
  const size_t SF  = (size_t)S_LEN * FFDIM;
  const size_t DF  = (size_t)DMODEL * FFDIM;

  float* ws   = (float*)d_ws;
  float* x    = ws;            ws += SD;
  float* x1   = ws;            ws += SD;
  float* tmp  = ws;            ws += SD;
  float* qb   = ws;            ws += SD;
  float* kb   = ws;            ws += SD;
  float* vb   = ws;            ws += SD;
  float* kgb  = ws;            ws += SD;
  float* vgb  = ws;            ws += SD;
  float* ctxb = ws;            ws += SD;
  float* hb   = ws;            ws += SF;
  float* qgb  = ws;            ws += DMODEL;
  unsigned short* xbf = (unsigned short*)ws;            // S*D bf16 (act or ctx)
  unsigned short* hbf = xbf + SD;                       // S*FF bf16
  unsigned short* wbf = hbf + SF;                       // up to D*FF bf16 weights

  const float SCALE = 0.125f;  // 1/sqrt(64)
  dim3 gD(DMODEL / 128, S_LEN / 128);
  dim3 gFF(FFDIM / 128, S_LEN / 128);
  const int cSD = (int)(SD / 2048), cSF = (int)(SF / 2048);
  const int cDD = (int)((size_t)DMODEL * DMODEL / 2048), cDF = (int)(DF / 2048);

  embed_ln<<<S_LEN, 256, 0, stream>>>(ids, ew, ep, eg, eb, x);

  for (int l = 0; l < NLAYER; ++l) {
    size_t o2 = (size_t)l * DMODEL * DMODEL;
    size_t o1 = (size_t)l * DMODEL;
    size_t oF = (size_t)l * DMODEL * FFDIM;
    size_t oFb = (size_t)l * FFDIM;

    cvt_bf16<<<cSD, 256, 0, stream>>>(x, xbf, (int)SD);
    cvt_bf16<<<cDD, 256, 0, stream>>>(Wq + o2, wbf, (int)(SD ? DMODEL * DMODEL : 0));
    gemm_bf16<<<gD, 512, 0, stream>>>(xbf, wbf, bq + o1, qb, S_LEN, DMODEL, DMODEL, SCALE, 0);
    cvt_bf16<<<cDD, 256, 0, stream>>>(Wk + o2, wbf, DMODEL * DMODEL);
    gemm_bf16<<<gD, 512, 0, stream>>>(xbf, wbf, bk + o1, kb, S_LEN, DMODEL, DMODEL, 1.f, 0);
    cvt_bf16<<<cDD, 256, 0, stream>>>(Wv + o2, wbf, DMODEL * DMODEL);
    gemm_bf16<<<gD, 512, 0, stream>>>(xbf, wbf, bv + o1, vb, S_LEN, DMODEL, DMODEL, 1.f, 0);
    cvt_bf16<<<cDD, 256, 0, stream>>>(Wkg + o2, wbf, DMODEL * DMODEL);
    gemm_bf16<<<gD, 512, 0, stream>>>(xbf, wbf, bkg + o1, kgb, S_LEN, DMODEL, DMODEL, 1.f, 0);
    cvt_bf16<<<cDD, 256, 0, stream>>>(Wvg + o2, wbf, DMODEL * DMODEL);
    gemm_bf16<<<gD, 512, 0, stream>>>(xbf, wbf, bvg + o1, vgb, S_LEN, DMODEL, DMODEL, 1.f, 0);
    vec_gemm<<<3, 256, 0, stream>>>(x, Wqg + o2, bqg + o1, qgb, DMODEL, SCALE);

    attn_window<<<dim3(NBLK, NHEAD), 256, 0, stream>>>(qb, kb, vb, mask, ctxb);
    attn_global<<<NHEAD, 256, 0, stream>>>(qgb, kgb, vgb, mask, ctxb);

    cvt_bf16<<<cSD, 256, 0, stream>>>(ctxb, xbf, (int)SD);
    cvt_bf16<<<cDD, 256, 0, stream>>>(Wo + o2, wbf, DMODEL * DMODEL);
    gemm_bf16<<<gD, 512, 0, stream>>>(xbf, wbf, bo + o1, tmp, S_LEN, DMODEL, DMODEL, 1.f, 0);
    add_ln<<<S_LEN, 256, 0, stream>>>(x, tmp, l1g + o1, l1b + o1, x1);

    cvt_bf16<<<cSD, 256, 0, stream>>>(x1, xbf, (int)SD);
    cvt_bf16<<<cDF, 256, 0, stream>>>(W1 + oF, wbf, (int)DF);
    gemm_bf16<<<gFF, 512, 0, stream>>>(xbf, wbf, b1 + oFb, hb, S_LEN, FFDIM, DMODEL, 1.f, 1);
    cvt_bf16<<<cSF, 256, 0, stream>>>(hb, hbf, (int)SF);
    cvt_bf16<<<cDF, 256, 0, stream>>>(W2 + oF, wbf, (int)DF);
    gemm_bf16<<<gD, 512, 0, stream>>>(hbf, wbf, b2 + o1, tmp, S_LEN, DMODEL, FFDIM, 1.f, 0);
    add_ln<<<S_LEN, 256, 0, stream>>>(x1, tmp, l2g + o1, l2b + o1, x);
  }

  classifier<<<1, 256, 0, stream>>>(x, Wc, bc, (float*)d_out);
}
